// TransformerBlock_1245540516051
// MI455X (gfx1250) — compile-verified, matrix operand order checked by reading, not run on hardware
//
#include <hip/hip_runtime.h>
#include <hip/hip_bf16.h>

// ---------------------------------------------------------------------------
// Transformer block forward for MI455X (gfx1250, wave32, WMMA).
// All GEMMs + attention matmuls run on v_wmma_f32_16x16x32_bf16 (fp32 accum).
// GEMM: 128x128 WG tile, 64x32 per wave (4x2 WMMA tiles), double-buffered LDS
// fed by GLOBAL_LOAD_ASYNC_TO_LDS_B128 (ASYNCcnt pipelining, CDNA5 path).
// Flash-style attention: scores never leave the WGP (LDS/registers only).
// ---------------------------------------------------------------------------

#define DEVINL __device__ __forceinline__

typedef __attribute__((ext_vector_type(16))) __bf16 v16bf;
typedef __attribute__((ext_vector_type(8)))  __bf16 v8bf;
typedef __attribute__((ext_vector_type(8)))  float  v8f;

union Frag { v16bf v; v8bf h[2]; };
union Pack4 { uint2 u; __bf16 b[4]; };

constexpr int Bc = 2, Tc = 2048, Ec = 1024, Hc = 16, Dc = 64, FFc = 4096;
constexpr int Mc = Bc * Tc;           // 4096 token rows

DEVINL __bf16 f2bf(float x) {
  unsigned u = __builtin_bit_cast(unsigned, x);
  unsigned r = (u + 0x7FFFu + ((u >> 16) & 1u)) >> 16;
  unsigned short s = (unsigned short)r;
  return __builtin_bit_cast(__bf16, s);
}

DEVINL v8f wmma_bf16(const Frag& a, const Frag& b, v8f c) {
  return __builtin_amdgcn_wmma_f32_16x16x32_bf16(
      /*neg_a=*/false, a.v, /*neg_b=*/false, b.v,
      /*c_mod=*/(short)0, c, /*reuse_a=*/false, /*reuse_b=*/false);
}

DEVINL float gelu_tanh(float v) {
  float u = 0.7978845608028654f * (v + 0.044715f * v * v * v);
  return 0.5f * v * (1.0f + tanhf(u));
}

// CDNA5 async global->LDS copy, 16B per lane, tracked by ASYNCcnt.
// GV addressing: lds_byte_addr (v), 64-bit global addr (v pair), no saddr.
DEVINL void async_g2l_b128(unsigned lds_addr, const void* gaddr) {
  asm volatile("global_load_async_to_lds_b128 %0, %1, off"
               :
               : "v"(lds_addr), "v"((unsigned long long)gaddr)
               : "memory");
}
DEVINL void wait_async4() { asm volatile("s_wait_asynccnt 4" ::: "memory"); }
DEVINL void wait_async0() { asm volatile("s_wait_asynccnt 0" ::: "memory"); }

// ---------------------------------------------------------------------------
// fp32 -> bf16 elementwise convert (weights), 4-wide packed stores
// ---------------------------------------------------------------------------
__global__ void cvt_f32_bf16(const float* __restrict__ in,
                             __bf16* __restrict__ out, int n4) {
  for (int i = blockIdx.x * blockDim.x + threadIdx.x; i < n4;
       i += gridDim.x * blockDim.x) {
    float4 v = ((const float4*)in)[i];
    Pack4 p;
    p.b[0] = f2bf(v.x); p.b[1] = f2bf(v.y);
    p.b[2] = f2bf(v.z); p.b[3] = f2bf(v.w);
    ((uint2*)out)[i] = p.u;
  }
}

// ---------------------------------------------------------------------------
// LayerNorm over E=1024, one block (256 thr) per row, packed bf16 output
// ---------------------------------------------------------------------------
__global__ __launch_bounds__(256) void ln_bf16(const float* __restrict__ x,
                                               const float* __restrict__ g,
                                               const float* __restrict__ bta,
                                               __bf16* __restrict__ out) {
  const int row  = blockIdx.x;
  const int tid  = threadIdx.x;
  const int lane = tid & 31, wave = tid >> 5;
  const float* xr = x + (size_t)row * Ec;

  float4 v = ((const float4*)xr)[tid];
  float sum = v.x + v.y + v.z + v.w;
  float ss  = v.x * v.x + v.y * v.y + v.z * v.z + v.w * v.w;
#pragma unroll
  for (int m = 1; m < 32; m <<= 1) {
    sum += __shfl_xor(sum, m, 32);
    ss  += __shfl_xor(ss,  m, 32);
  }
  __shared__ float s1[8], s2[8];
  if (lane == 0) { s1[wave] = sum; s2[wave] = ss; }
  __syncthreads();
  if (wave == 0) {
    float a = (lane < 8) ? s1[lane] : 0.f;
    float c = (lane < 8) ? s2[lane] : 0.f;
#pragma unroll
    for (int m = 1; m < 8; m <<= 1) {
      a += __shfl_xor(a, m, 32);
      c += __shfl_xor(c, m, 32);
    }
    if (lane == 0) { s1[0] = a; s2[0] = c; }
  }
  __syncthreads();
  const float mu  = s1[0] * (1.0f / Ec);
  const float var = s2[0] * (1.0f / Ec) - mu * mu;
  const float rv  = rsqrtf(var + 1e-5f);

  float4 gv = ((const float4*)g)[tid];
  float4 bv = ((const float4*)bta)[tid];
  Pack4 p;
  p.b[0] = f2bf((v.x - mu) * rv * gv.x + bv.x);
  p.b[1] = f2bf((v.y - mu) * rv * gv.y + bv.y);
  p.b[2] = f2bf((v.z - mu) * rv * gv.z + bv.z);
  p.b[3] = f2bf((v.w - mu) * rv * gv.w + bv.w);
  ((uint2*)(out + (size_t)row * Ec))[tid] = p.u;
}

// ---------------------------------------------------------------------------
// C(M,N) = A(M,K) @ W(N,K)^T + bias, optional GELU, optional residual
//   out = res + (*wscale) * C  (RES)      fp32 and/or bf16 stores
// Tiles: 128x128x32, 256 threads (8 waves), wave -> 64x32 (4x2 WMMA tiles).
// Double-buffered LDS fed by async global->LDS copies (ASYNCcnt pipeline).
// ---------------------------------------------------------------------------
template <bool GELU, bool RES>
__global__ __launch_bounds__(256) void gemm_bf16(
    const __bf16* __restrict__ A, const __bf16* __restrict__ W,
    const float* __restrict__ bias, const float* __restrict__ res,
    const float* __restrict__ wscale, float* __restrict__ outf,
    __bf16* __restrict__ outb, int Mdim, int Ndim, int Kdim) {
  constexpr int BK = 32, LDA = BK + 8;             // padded LDS stride
  constexpr unsigned BUFB = 128 * LDA * 2;         // bytes per buffer
  __shared__ __bf16 As[2 * 128 * LDA];
  __shared__ __bf16 Bs[2 * 128 * LDA];

  const int tid  = threadIdx.x;
  const int lane = tid & 31, wave = tid >> 5;
  const int fr = lane & 15, kh = (lane >> 4) * 8, hi = lane >> 4;
  const int m0 = blockIdx.y * 128, n0 = blockIdx.x * 128;
  const int wm = (wave & 1) * 64, wn = (wave >> 1) * 32;

  v8f acc[4][2];
#pragma unroll
  for (int i = 0; i < 4; ++i)
#pragma unroll
    for (int j = 0; j < 2; ++j)
#pragma unroll
      for (int r = 0; r < 8; ++r) acc[i][j][r] = 0.f;

  // per-thread copy slots: 4 x 16B per tile
  const int lrow = tid >> 2, lq = (tid & 3) * 8;
  const __bf16* ga0 = A + (size_t)(m0 + lrow) * Kdim + lq;
  const __bf16* ga1 = A + (size_t)(m0 + lrow + 64) * Kdim + lq;
  const __bf16* gb0 = W + (size_t)(n0 + lrow) * Kdim + lq;
  const __bf16* gb1 = W + (size_t)(n0 + lrow + 64) * Kdim + lq;
  const unsigned la0 = (unsigned)(size_t)As + (unsigned)(lrow * LDA + lq) * 2;
  const unsigned la1 = (unsigned)(size_t)As + (unsigned)((lrow + 64) * LDA + lq) * 2;
  const unsigned lb0 = (unsigned)(size_t)Bs + (unsigned)(lrow * LDA + lq) * 2;
  const unsigned lb1 = (unsigned)(size_t)Bs + (unsigned)((lrow + 64) * LDA + lq) * 2;

  auto issue_tile = [&](int k0, int buf) {
    const unsigned bo = (unsigned)buf * BUFB;
    async_g2l_b128(la0 + bo, ga0 + k0);
    async_g2l_b128(la1 + bo, ga1 + k0);
    async_g2l_b128(lb0 + bo, gb0 + k0);
    async_g2l_b128(lb1 + bo, gb1 + k0);
  };

  issue_tile(0, 0);
  const int nt = Kdim / BK;
  for (int it = 0; it < nt; ++it) {
    const int cur = it & 1;
    if (it + 1 < nt) {
      issue_tile((it + 1) * BK, cur ^ 1);
      wait_async4();                    // tile `it` landed in LDS (wave-local)
    } else {
      wait_async0();
    }
    __syncthreads();                    // all waves' tile `it` visible

    const __bf16* Ab = As + cur * 128 * LDA;
    const __bf16* Bb = Bs + cur * 128 * LDA;
    Frag fa[4], fb[2];
#pragma unroll
    for (int i = 0; i < 4; ++i) {
      const __bf16* p = Ab + (wm + i * 16 + fr) * LDA;
      fa[i].h[0] = *(const v8bf*)(p + kh);
      fa[i].h[1] = *(const v8bf*)(p + 16 + kh);
    }
#pragma unroll
    for (int j = 0; j < 2; ++j) {
      const __bf16* p = Bb + (wn + j * 16 + fr) * LDA;
      fb[j].h[0] = *(const v8bf*)(p + kh);
      fb[j].h[1] = *(const v8bf*)(p + 16 + kh);
    }
#pragma unroll
    for (int i = 0; i < 4; ++i)
#pragma unroll
      for (int j = 0; j < 2; ++j) acc[i][j] = wmma_bf16(fa[i], fb[j], acc[i][j]);
    __syncthreads();                    // reads done before buffer reuse
  }

  const float w = RES ? *wscale : 0.f;
#pragma unroll
  for (int i = 0; i < 4; ++i)
#pragma unroll
    for (int j = 0; j < 2; ++j)
#pragma unroll
      for (int r = 0; r < 8; ++r) {
        const int m = m0 + wm + i * 16 + r + hi * 8;
        const int n = n0 + wn + j * 16 + fr;
        float v = acc[i][j][r] + bias[n];
        if (GELU) v = gelu_tanh(v);
        if (RES)  v = res[(size_t)m * Ndim + n] + w * v;
        if (outf) outf[(size_t)m * Ndim + n] = v;
        if (outb) outb[(size_t)m * Ndim + n] = f2bf(v);
      }
}

// ---------------------------------------------------------------------------
// Flash attention. Grid: (T/128, B*H). 256 thr = 8 waves; wave = 16 q rows.
// qkv fp32 (M, 3E); output y bf16 (M, E).
// ---------------------------------------------------------------------------
__global__ __launch_bounds__(256) void attn_flash(const float* __restrict__ qkv,
                                                  __bf16* __restrict__ ybf) {
  constexpr int LDK = Dc + 8;                     // 72, padded
  __shared__ __bf16 Kl[64 * LDK];                 // K tile   [t][d]
  __shared__ __bf16 Vt[64 * LDK];                 // V tile^T [d][t]
  __shared__ __bf16 Pl[8 * 16 * LDK];             // per-wave P staging

  const int tid  = threadIdx.x;
  const int lane = tid & 31, wave = tid >> 5;
  const int fr = lane & 15, kh = (lane >> 4) * 8, hi = lane >> 4;
  const int b = blockIdx.y / Hc, h = blockIdx.y % Hc;
  const int qrow0 = blockIdx.x * 128 + wave * 16;

  // Q fragments (16 rows x 64 d) held in registers, converted to bf16
  Frag a0, a1;
  {
    const int t = qrow0 + fr;
    const float* qp = qkv + ((size_t)(b * Tc + t) * 3) * Ec + h * Dc;
#pragma unroll
    for (int e = 0; e < 8; ++e) {
      a0.v[e]     = f2bf(qp[kh + e]);
      a0.v[8 + e] = f2bf(qp[16 + kh + e]);
      a1.v[e]     = f2bf(qp[32 + kh + e]);
      a1.v[8 + e] = f2bf(qp[48 + kh + e]);
    }
  }

  v8f o_acc[4];
#pragma unroll
  for (int j = 0; j < 4; ++j)
#pragma unroll
    for (int r = 0; r < 8; ++r) o_acc[j][r] = 0.f;
  float m_i[8], l_i[8];
#pragma unroll
  for (int r = 0; r < 8; ++r) { m_i[r] = -1e30f; l_i[r] = 0.f; }

  const float scale = 0.125f;                     // 1/sqrt(64)
  const int lrow = tid >> 2, lseg = (tid & 3) * 16;   // K-tile mapping
  const int tb = (tid >> 4) * 4, db = (tid & 15) * 4; // V-tile 4x4 blocks
  __bf16* pw = Pl + wave * 16 * LDK;

  for (int kv0 = 0; kv0 < Tc; kv0 += 64) {
    __syncthreads();
    {  // K tile: rows of 16, packed 8B LDS stores
      const float* kp =
          qkv + ((size_t)(b * Tc + kv0 + lrow) * 3 + 1) * Ec + h * Dc + lseg;
#pragma unroll
      for (int e = 0; e < 16; e += 4) {
        float4 k4 = *(const float4*)(kp + e);
        Pack4 pk;
        pk.b[0] = f2bf(k4.x); pk.b[1] = f2bf(k4.y);
        pk.b[2] = f2bf(k4.z); pk.b[3] = f2bf(k4.w);
        *(uint2*)(Kl + lrow * LDK + lseg + e) = pk.u;
      }
    }
    {  // V^T tile: 4x4 register transpose, packed 8B LDS stores
      float mat[4][4];
#pragma unroll
      for (int i = 0; i < 4; ++i) {
        const float* vp =
            qkv + ((size_t)(b * Tc + kv0 + tb + i) * 3 + 2) * Ec + h * Dc + db;
        float4 v4 = *(const float4*)vp;
        mat[i][0] = v4.x; mat[i][1] = v4.y; mat[i][2] = v4.z; mat[i][3] = v4.w;
      }
#pragma unroll
      for (int j = 0; j < 4; ++j) {
        Pack4 pk;
        pk.b[0] = f2bf(mat[0][j]); pk.b[1] = f2bf(mat[1][j]);
        pk.b[2] = f2bf(mat[2][j]); pk.b[3] = f2bf(mat[3][j]);
        *(uint2*)(Vt + (db + j) * LDK + tb) = pk.u;
      }
    }
    __syncthreads();

    // S = Q K^T  (16 x 64)
    v8f s[4];
#pragma unroll
    for (int j = 0; j < 4; ++j) {
      const __bf16* p = Kl + (j * 16 + fr) * LDK;
      Frag b0, b1;
      b0.h[0] = *(const v8bf*)(p + kh);      b0.h[1] = *(const v8bf*)(p + 16 + kh);
      b1.h[0] = *(const v8bf*)(p + 32 + kh); b1.h[1] = *(const v8bf*)(p + 48 + kh);
      v8f z;
#pragma unroll
      for (int r = 0; r < 8; ++r) z[r] = 0.f;
      z = wmma_bf16(a0, b0, z);
      z = wmma_bf16(a1, b1, z);
      s[j] = z;
    }

    // online softmax update (row stats replicated across each 16-lane half)
#pragma unroll
    for (int r = 0; r < 8; ++r) {
      float mx = -1e30f;
#pragma unroll
      for (int j = 0; j < 4; ++j) {
        s[j][r] *= scale;
        mx = fmaxf(mx, s[j][r]);
      }
#pragma unroll
      for (int m = 1; m < 16; m <<= 1) mx = fmaxf(mx, __shfl_xor(mx, m, 32));
      const float newm = fmaxf(m_i[r], mx);
      const float f = __expf(m_i[r] - newm);
      float sum = 0.f;
#pragma unroll
      for (int j = 0; j < 4; ++j) {
        float p = __expf(s[j][r] - newm);
        s[j][r] = p;
        sum += p;
      }
#pragma unroll
      for (int m = 1; m < 16; m <<= 1) sum += __shfl_xor(sum, m, 32);
      l_i[r] = l_i[r] * f + sum;
      m_i[r] = newm;
#pragma unroll
      for (int j = 0; j < 4; ++j) o_acc[j][r] *= f;
    }

    // P: C-layout -> A-fragment layout via per-wave LDS staging
#pragma unroll
    for (int j = 0; j < 4; ++j)
#pragma unroll
      for (int r = 0; r < 8; ++r)
        pw[(r + hi * 8) * LDK + j * 16 + fr] = f2bf(s[j][r]);

    Frag p0, p1;
    {
      const __bf16* p = pw + fr * LDK;
      p0.h[0] = *(const v8bf*)(p + kh);      p0.h[1] = *(const v8bf*)(p + 16 + kh);
      p1.h[0] = *(const v8bf*)(p + 32 + kh); p1.h[1] = *(const v8bf*)(p + 48 + kh);
    }

    // O += P V
#pragma unroll
    for (int j = 0; j < 4; ++j) {
      const __bf16* p = Vt + (j * 16 + fr) * LDK;
      Frag bv0, bv1;
      bv0.h[0] = *(const v8bf*)(p + kh);      bv0.h[1] = *(const v8bf*)(p + 16 + kh);
      bv1.h[0] = *(const v8bf*)(p + 32 + kh); bv1.h[1] = *(const v8bf*)(p + 48 + kh);
      o_acc[j] = wmma_bf16(p0, bv0, o_acc[j]);
      o_acc[j] = wmma_bf16(p1, bv1, o_acc[j]);
    }
  }

  // normalize and emit bf16 y
#pragma unroll
  for (int r = 0; r < 8; ++r) {
    const float inv = 1.0f / l_i[r];
    const int t = qrow0 + r + hi * 8;
    __bf16* yp = ybf + (size_t)(b * Tc + t) * Ec + h * Dc;
#pragma unroll
    for (int j = 0; j < 4; ++j) yp[j * 16 + fr] = f2bf(o_acc[j][r] * inv);
  }
}

// ---------------------------------------------------------------------------
// Host-side orchestration
// ---------------------------------------------------------------------------
extern "C" void kernel_launch(void* const* d_in, const int* in_sizes, int n_in,
                              void* d_out, int out_size, void* d_ws,
                              size_t ws_size, hipStream_t stream) {
  const float* x     = (const float*)d_in[0];
  const float* ln1_g = (const float*)d_in[1];
  const float* ln1_b = (const float*)d_in[2];
  const float* Wqkv  = (const float*)d_in[3];
  const float* bqkv  = (const float*)d_in[4];
  const float* Wo    = (const float*)d_in[5];
  const float* bo    = (const float*)d_in[6];
  const float* w_att = (const float*)d_in[7];
  const float* ln2_g = (const float*)d_in[8];
  const float* ln2_b = (const float*)d_in[9];
  const float* W1    = (const float*)d_in[10];
  const float* b1    = (const float*)d_in[11];
  const float* W2    = (const float*)d_in[12];
  const float* b2    = (const float*)d_in[13];
  const float* w_mlp = (const float*)d_in[14];
  float* out = (float*)d_out;

  // bump allocator over workspace
  char* ws = (char*)d_ws;
  auto alloc = [&](size_t bytes) -> char* {
    char* p = ws;
    ws += (bytes + 255) & ~(size_t)255;
    return p;
  };
  __bf16* Wqkv_bf = (__bf16*)alloc((size_t)3 * Ec * Ec * 2);
  __bf16* Wo_bf   = (__bf16*)alloc((size_t)Ec * Ec * 2);
  __bf16* W1_bf   = (__bf16*)alloc((size_t)FFc * Ec * 2);
  __bf16* W2_bf   = (__bf16*)alloc((size_t)Ec * FFc * 2);
  __bf16* h_bf    = (__bf16*)alloc((size_t)Mc * Ec * 2);      // LN1 / LN2 out
  float*  qkv_f   = (float*)alloc((size_t)Mc * 3 * Ec * 4);   // reused as g_bf
  __bf16* y_bf    = (__bf16*)alloc((size_t)Mc * Ec * 2);
  float*  x1      = (float*)alloc((size_t)Mc * Ec * 4);
  __bf16* g_bf    = (__bf16*)qkv_f;  // GELU activations reuse qkv region

  // 1) weights -> bf16 (packed 4-wide)
  cvt_f32_bf16<<<1024, 256, 0, stream>>>(Wqkv, Wqkv_bf, 3 * Ec * Ec / 4);
  cvt_f32_bf16<<<512,  256, 0, stream>>>(Wo, Wo_bf, Ec * Ec / 4);
  cvt_f32_bf16<<<1024, 256, 0, stream>>>(W1, W1_bf, FFc * Ec / 4);
  cvt_f32_bf16<<<1024, 256, 0, stream>>>(W2, W2_bf, Ec * FFc / 4);

  // 2) h = LN1(x) in bf16
  ln_bf16<<<Mc, 256, 0, stream>>>(x, ln1_g, ln1_b, h_bf);

  // 3) qkv = h @ Wqkv^T + bqkv  (fp32)
  gemm_bf16<false, false><<<dim3(3 * Ec / 128, Mc / 128), 256, 0, stream>>>(
      h_bf, Wqkv_bf, bqkv, nullptr, nullptr, qkv_f, nullptr, Mc, 3 * Ec, Ec);

  // 4) flash attention -> y (bf16)
  attn_flash<<<dim3(Tc / 128, Bc * Hc), 256, 0, stream>>>(qkv_f, y_bf);

  // 5) x1 = x + w_att * (y @ Wo^T + bo)
  gemm_bf16<false, true><<<dim3(Ec / 128, Mc / 128), 256, 0, stream>>>(
      y_bf, Wo_bf, bo, x, w_att, x1, nullptr, Mc, Ec, Ec);

  // 6) h2 = LN2(x1) in bf16 (reuse h_bf)
  ln_bf16<<<Mc, 256, 0, stream>>>(x1, ln2_g, ln2_b, h_bf);

  // 7) g = gelu(h2 @ W1^T + b1) in bf16 (reuses qkv region)
  gemm_bf16<true, false><<<dim3(FFc / 128, Mc / 128), 256, 0, stream>>>(
      h_bf, W1_bf, b1, nullptr, nullptr, nullptr, g_bf, Mc, FFc, Ec);

  // 8) out = x1 + w_mlp * (g @ W2^T + b2)
  gemm_bf16<false, true><<<dim3(Ec / 128, Mc / 128), 256, 0, stream>>>(
      g_bf, W2_bf, b2, x1, w_mlp, out, nullptr, Mc, Ec, FFc);
}
